// Mask_17695265259591
// MI455X (gfx1250) — matrix operand order repaired and checked by reading, not compiled, not run on hardware
//
#include <hip/hip_runtime.h>
#include <stdint.h>

// ---------------------------------------------------------------------------
// Supermasked 3-layer MLP for MI455X (gfx1250, wave32, WMMA)
//   thr   = radix-select median of all 50.3M scores (4x 8-bit passes)
//   Wb    = bf16( w * (s >= thr) ), biasm = b * (s_b >= thr)
//   out   = relu(relu(x Wb1^T + b1) Wb2^T + b2) Wb3^T + b3
// GEMMs use v_wmma_f32_16x16x32_bf16 with double-buffered async LDS staging.
// ---------------------------------------------------------------------------

typedef __attribute__((ext_vector_type(16))) __bf16 v16bf;
typedef __attribute__((ext_vector_type(8)))  float  v8f;
typedef __attribute__((ext_vector_type(4)))  int    v4i;

#define BM 128
#define BN 128
#define BK 32
#define LDSPITCH 48   // halves per LDS row: 32 data + 16 pad (96B, 32B-aligned)

#if defined(__gfx1250__) && __has_builtin(__builtin_amdgcn_global_load_async_to_lds_b128) && __has_builtin(__builtin_amdgcn_s_wait_asynccnt)
#define USE_ASYNC_LDS 1
#else
#define USE_ASYNC_LDS 0
#endif

__device__ __forceinline__ void copy16(const __bf16* g, __bf16* l) {
#if USE_ASYNC_LDS
  __builtin_amdgcn_global_load_async_to_lds_b128(
      (__attribute__((address_space(1))) v4i*)(g),
      (__attribute__((address_space(3))) v4i*)(l),
      0, 0);
#else
  *reinterpret_cast<int4*>(l) = *reinterpret_cast<const int4*>(g);
#endif
}

#if USE_ASYNC_LDS
#define ASYNC_WAIT(n) __builtin_amdgcn_s_wait_asynccnt(n)
#else
#define ASYNC_WAIT(n)
#endif

// ------------------------- radix select (median) ---------------------------

__device__ __forceinline__ unsigned f2key(float f) {
  unsigned u = __float_as_uint(f);
  return (u & 0x80000000u) ? ~u : (u | 0x80000000u);
}

__global__ void init_select(unsigned* hist, unsigned* state, unsigned rank0) {
  if (threadIdx.x == 0) { state[0] = 0u; state[1] = rank0; state[2] = 0u; }
  hist[threadIdx.x] = 0u;   // launched with 256 threads
}

__global__ void hist_pass(const float* __restrict__ s, long n, unsigned shift,
                          const unsigned* __restrict__ state, unsigned* __restrict__ hist) {
  __shared__ unsigned lh[256];
  lh[threadIdx.x] = 0u;
  __syncthreads();
  const unsigned prefix = state[0];
  const unsigned sh2 = shift + 8u;
  const long stride = (long)gridDim.x * blockDim.x;
  for (long i = (long)blockIdx.x * blockDim.x + threadIdx.x; i < n; i += stride) {
    unsigned key = f2key(s[i]);
    bool ok = (sh2 >= 32u) || (((key ^ prefix) >> sh2) == 0u);
    if (ok) atomicAdd(&lh[(key >> shift) & 0xFFu], 1u);
  }
  __syncthreads();
  unsigned c = lh[threadIdx.x];
  if (c) atomicAdd(&hist[threadIdx.x], c);
}

__global__ void scan_pass(unsigned* hist, unsigned* state, unsigned shift, int last) {
  if (threadIdx.x == 0) {
    unsigned rank = state[1];
    unsigned prefix = state[0];
    unsigned cum = 0u, b = 255u;
    for (unsigned i = 0; i < 256u; ++i) {
      unsigned c = hist[i];
      if (cum + c > rank) { b = i; break; }
      cum += c;
    }
    prefix |= (b << shift);
    state[0] = prefix;
    state[1] = rank - cum;
    if (last) {   // invert sortable-key map -> threshold float bits
      unsigned u = (prefix & 0x80000000u) ? (prefix ^ 0x80000000u) : ~prefix;
      state[2] = u;
    }
  }
  __syncthreads();
  hist[threadIdx.x] = 0u;   // ready for next pass (256 threads)
}

// ------------------------- masking / conversion ----------------------------

__global__ void mask_weights(const float4* __restrict__ w, const float4* __restrict__ s,
                             __bf16* __restrict__ out, long n4,
                             const unsigned* __restrict__ state) {
  const float thr = __uint_as_float(state[2]);
  const long stride = (long)gridDim.x * blockDim.x;
  for (long i = (long)blockIdx.x * blockDim.x + threadIdx.x; i < n4; i += stride) {
    float4 wv = w[i];
    float4 sv = s[i];
    __bf16* o = out + i * 4;
    o[0] = (sv.x >= thr) ? (__bf16)wv.x : (__bf16)0.0f;
    o[1] = (sv.y >= thr) ? (__bf16)wv.y : (__bf16)0.0f;
    o[2] = (sv.z >= thr) ? (__bf16)wv.z : (__bf16)0.0f;
    o[3] = (sv.w >= thr) ? (__bf16)wv.w : (__bf16)0.0f;
  }
}

__global__ void mask_bias(const float* b1, const float* s1,
                          const float* b2, const float* s2,
                          const float* b3, const float* s3,
                          float* out, int d, const unsigned* state) {
  const float thr = __uint_as_float(state[2]);
  int i = blockIdx.x * blockDim.x + threadIdx.x;
  if (i < d) {
    out[i]         = (s1[i] >= thr) ? b1[i] : 0.0f;
    out[d + i]     = (s2[i] >= thr) ? b2[i] : 0.0f;
    out[2 * d + i] = (s3[i] >= thr) ? b3[i] : 0.0f;
  }
}

__global__ void convert_x(const float4* __restrict__ x, __bf16* __restrict__ out, long n4) {
  const long stride = (long)gridDim.x * blockDim.x;
  for (long i = (long)blockIdx.x * blockDim.x + threadIdx.x; i < n4; i += stride) {
    float4 v = x[i];
    __bf16* o = out + i * 4;
    o[0] = (__bf16)v.x; o[1] = (__bf16)v.y; o[2] = (__bf16)v.z; o[3] = (__bf16)v.w;
  }
}

// ------------------------------ WMMA GEMM ----------------------------------
// C[M,N] = A[M,K] * W[N,K]^T  (+ bias, optional relu)
// Block: 128x128 tile, 256 threads = 8 wave32 waves (4 row-waves x 2 col-waves)
// Wave: 32 rows x 64 cols -> 2x4 fragments of 16x16, K stepped by 32.

__global__ __launch_bounds__(256) void mlp_gemm_wmma(
    const __bf16* __restrict__ A,    // [M][K]
    const __bf16* __restrict__ W,    // [N][K] (row-major torch weight)
    const float*  __restrict__ bias, // [N] pre-masked
    __bf16* __restrict__ outB,       // [M][N] bf16 (hidden layers) or null
    float*  __restrict__ outF,       // [M][N] f32 (last layer) or null
    int M, int N, int K, int relu)
{
  __shared__ __bf16 As[2][BM * LDSPITCH];
  __shared__ __bf16 Bs[2][BN * LDSPITCH];

  const int tid  = threadIdx.x;
  const int lane = tid & 31;
  const int wv   = tid >> 5;      // 0..7
  const int wrow = wv & 3;        // 4 row-waves  -> 32 rows each
  const int wcol = wv >> 2;       // 2 col-waves  -> 64 cols each
  const int lm   = lane & 15;     // position 0..15 within fragment
  const int lh   = lane >> 4;     // half-wave select

  const int mBase = blockIdx.y * BM;
  const int nBase = blockIdx.x * BN;
  const int T = K / BK;

  // Stage one 128x32 A tile and one 128x32 W tile (4 x b128 per thread).
  auto copy_tile = [&](int buf, int t) {
    const int kb = t * BK;
#pragma unroll
    for (int i = 0; i < 2; ++i) {
      int c   = tid + i * 256;    // 0..511 chunks of 16B (8 halves)
      int row = c >> 2;           // 0..127
      int ch  = (c & 3) * 8;      // half offset within row
      copy16(&A[(size_t)(mBase + row) * K + kb + ch], &As[buf][row * LDSPITCH + ch]);
      copy16(&W[(size_t)(nBase + row) * K + kb + ch], &Bs[buf][row * LDSPITCH + ch]);
    }
  };

  v8f acc[2][4] = {};

  copy_tile(0, 0);
  for (int t = 0; t < T; ++t) {
    const int cur = t & 1;
    if (t + 1 < T) { copy_tile(cur ^ 1, t + 1); ASYNC_WAIT(4); }  // keep only next tile in flight
    else           { ASYNC_WAIT(0); }
    __syncthreads();

    // A fragment: 16x32 bf16, ISA layout: lanes 0-15 -> K {0..7,16..23},
    // lanes 16-31 -> K {8..15,24..31}, pairs packed per VGPR.
    v16bf fa[2];
#pragma unroll
    for (int i = 0; i < 2; ++i) {
      const __bf16* ar = &As[cur][(wrow * 32 + i * 16 + lm) * LDSPITCH];
      const int kb = lh * 8;
#pragma unroll
      for (int e = 0; e < 8; ++e)  fa[i][e] = ar[kb + e];
#pragma unroll
      for (int e = 8; e < 16; ++e) fa[i][e] = ar[kb + e + 8];
    }
    // B fragment: 32x16 bf16, lanes 0-15 -> K {0..7,16..23}, lanes 16-31 -> K {8..15,24..31};
    // B[k][n] = W[n][k] so each lane reads halves of one W row.
    v16bf fb[4];
#pragma unroll
    for (int j = 0; j < 4; ++j) {
      const __bf16* br = &Bs[cur][(wcol * 64 + j * 16 + lm) * LDSPITCH];
      const int kb = lh * 8;
#pragma unroll
      for (int e = 0; e < 8; ++e)  fb[j][e] = br[kb + e];
#pragma unroll
      for (int e = 8; e < 16; ++e) fb[j][e] = br[kb + e + 8];
    }

#pragma unroll
    for (int i = 0; i < 2; ++i)
#pragma unroll
      for (int j = 0; j < 4; ++j)
        acc[i][j] = __builtin_amdgcn_wmma_f32_16x16x32_bf16(
            false, fa[i], false, fb[j], (short)0, acc[i][j], false, false);

    __syncthreads();
  }

  // Epilogue: C/D layout: VGPR v -> M = v (lanes 0-15) / v+8 (lanes 16-31), N = lane%16
#pragma unroll
  for (int j = 0; j < 4; ++j) {
    const int gcol = nBase + wcol * 64 + j * 16 + lm;
    const float bv = bias[gcol];
#pragma unroll
    for (int i = 0; i < 2; ++i) {
#pragma unroll
      for (int v = 0; v < 8; ++v) {
        const int grow = mBase + wrow * 32 + i * 16 + lh * 8 + v;
        float val = acc[i][j][v] + bv;
        if (relu) val = fmaxf(val, 0.0f);
        if (outF) outF[(size_t)grow * N + gcol] = val;
        else      outB[(size_t)grow * N + gcol] = (__bf16)val;
      }
    }
  }
}

// ------------------------------- host side ---------------------------------

extern "C" void kernel_launch(void* const* d_in, const int* in_sizes, int n_in,
                              void* d_out, int out_size, void* d_ws, size_t ws_size,
                              hipStream_t stream) {
  const float* x   = (const float*)d_in[0];
  const float* w1  = (const float*)d_in[1];
  const float* b1  = (const float*)d_in[2];
  const float* w2  = (const float*)d_in[3];
  const float* b2  = (const float*)d_in[4];
  const float* w3  = (const float*)d_in[5];
  const float* b3  = (const float*)d_in[6];
  const float* sw1 = (const float*)d_in[7];
  const float* sb1 = (const float*)d_in[8];
  const float* sw2 = (const float*)d_in[9];
  const float* sb2 = (const float*)d_in[10];
  const float* sw3 = (const float*)d_in[11];
  const float* sb3 = (const float*)d_in[12];
  (void)n_in; (void)out_size; (void)ws_size;

  const int Dd = 4096, Bb = 2048;

  uint8_t* ws = (uint8_t*)d_ws;
  unsigned* hist  = (unsigned*)(ws);            // 256 * 4B
  unsigned* state = (unsigned*)(ws + 1024);     // {prefix, rank, thr_bits}
  float*    biasm = (float*)(ws + 4096);        // 3 * 4096 f32
  size_t o = 65536;
  __bf16* xb  = (__bf16*)(ws + o); o += (size_t)Bb * Dd * 2;
  __bf16* w1b = (__bf16*)(ws + o); o += (size_t)Dd * Dd * 2;
  __bf16* w2b = (__bf16*)(ws + o); o += (size_t)Dd * Dd * 2;
  __bf16* w3b = (__bf16*)(ws + o); o += (size_t)Dd * Dd * 2;
  __bf16* h1  = (__bf16*)(ws + o); o += (size_t)Bb * Dd * 2;
  __bf16* h2  = (__bf16*)(ws + o); o += (size_t)Bb * Dd * 2;

  // global top-k threshold rank: j = int((1 - 0.5) * total)
  size_t total = 0;
  for (int i = 7; i < 13; ++i) total += (size_t)in_sizes[i];
  unsigned rank0 = (unsigned)(total / 2);

  init_select<<<1, 256, 0, stream>>>(hist, state, rank0);

  const float* scores[6] = {sw1, sb1, sw2, sb2, sw3, sb3};
  for (int pass = 0; pass < 4; ++pass) {
    unsigned shift = 24u - 8u * (unsigned)pass;
    for (int k = 0; k < 6; ++k) {
      long n = (long)in_sizes[7 + k];
      int blocks = (int)((n + 255) / 256);
      if (blocks > 4096) blocks = 4096;
      hist_pass<<<blocks, 256, 0, stream>>>(scores[k], n, shift, state, hist);
    }
    scan_pass<<<1, 256, 0, stream>>>(hist, state, shift, pass == 3);
  }

  const long nw4 = (long)Dd * Dd / 4;
  mask_weights<<<4096, 256, 0, stream>>>((const float4*)w1, (const float4*)sw1, w1b, nw4, state);
  mask_weights<<<4096, 256, 0, stream>>>((const float4*)w2, (const float4*)sw2, w2b, nw4, state);
  mask_weights<<<4096, 256, 0, stream>>>((const float4*)w3, (const float4*)sw3, w3b, nw4, state);
  mask_bias<<<(Dd + 255) / 256, 256, 0, stream>>>(b1, sb1, b2, sb2, b3, sb3, biasm, Dd, state);
  convert_x<<<2048, 256, 0, stream>>>((const float4*)x, xb, (long)Bb * Dd / 4);

  dim3 grid(Dd / BN, Bb / BM);   // (32, 16)
  mlp_gemm_wmma<<<grid, 256, 0, stream>>>(xb, w1b, biasm,          h1, nullptr, Bb, Dd, Dd, 1);
  mlp_gemm_wmma<<<grid, 256, 0, stream>>>(h1, w2b, biasm + Dd,     h2, nullptr, Bb, Dd, Dd, 1);
  mlp_gemm_wmma<<<grid, 256, 0, stream>>>(h2, w3b, biasm + 2 * Dd, nullptr, (float*)d_out, Bb, Dd, Dd, 0);
}
